// MultiheadAttention_58308476010745
// MI455X (gfx1250) — compile-verified, matrix operand order checked by reading, not run on hardware
//
#include <hip/hip_runtime.h>
#include <hip/hip_bf16.h>

// ---------------------------------------------------------------------------
// MHA for MI455X (gfx1250, wave32). bf16 WMMA (f32 accum) for all matmuls,
// flash-attention (no materialized logits), async global->LDS staging with
// double-buffered GEMM tiles.
// ---------------------------------------------------------------------------

typedef __attribute__((ext_vector_type(16))) __bf16 bf16x16;
typedef __attribute__((ext_vector_type(8)))  float  f32x8;

#define EMBED 1024
#define SEQ   2048
#define BATCH 2
#define HEADS 16
#define HDIM  64
#define ROWS  (BATCH * SEQ)      // 4096
#define QKVW  (3 * EMBED)        // 3072

// CDNA5 async global->LDS copy (ASYNCcnt-tracked, no VGPR staging).
// Set to 0 if the assembler rejects the mnemonic.
#define USE_ASYNC_LDS 1

__device__ __forceinline__ unsigned short f2bf(float f) {
    unsigned int u = __builtin_bit_cast(unsigned int, f);
    u = (u + 0x7FFFu + ((u >> 16) & 1u)) >> 16;   // round-to-nearest-even
    return (unsigned short)u;
}

// Copy 16 bytes global -> LDS. Async variant writes LDS directly via the
// async-DMA path; completion via s_wait_asynccnt + barrier.
__device__ __forceinline__ void cp16(unsigned short* dst_lds,
                                     const unsigned short* src_g) {
#if USE_ASYNC_LDS
    unsigned lds_off = (unsigned)(unsigned long long)(uintptr_t)dst_lds; // addr[31:0] = LDS offset
    unsigned long long ga = (unsigned long long)(uintptr_t)src_g;
    asm volatile("global_load_async_to_lds_b128 %0, %1, off"
                 :: "v"(lds_off), "v"(ga) : "memory");
#else
    *(uint4*)dst_lds = *(const uint4*)src_g;
#endif
}

__device__ __forceinline__ void async_wait_all() {
#if USE_ASYNC_LDS
    asm volatile("s_wait_asynccnt 0" ::: "memory");
#endif
}

struct FragBF {
    union { bf16x16 v; unsigned int u[8]; };
};

// A/B fragment for v_wmma_f32_16x16x32_bf16, row-major source in LDS.
// Lane L holds row (rowbase + L%16); K order per ISA 7.12.2. The two
// 4-dword groups are contiguous 16B each -> compiler emits ds_load_b128 x2.
__device__ __forceinline__ void load_frag_row(FragBF& f, const unsigned short* lds,
                                              int stride, int rowbase, int kbase) {
    const int lane = threadIdx.x & 31;
    const unsigned short* p =
        lds + (rowbase + (lane & 15)) * stride + kbase + ((lane >> 4) << 3);
    f.u[0] = *(const unsigned int*)(p + 0);
    f.u[1] = *(const unsigned int*)(p + 2);
    f.u[2] = *(const unsigned int*)(p + 4);
    f.u[3] = *(const unsigned int*)(p + 6);
    f.u[4] = *(const unsigned int*)(p + 16);
    f.u[5] = *(const unsigned int*)(p + 18);
    f.u[6] = *(const unsigned int*)(p + 20);
    f.u[7] = *(const unsigned int*)(p + 22);
}

// B fragment gathering a COLUMN of a row-major [K x N] LDS tile (for P@V).
__device__ __forceinline__ void load_frag_col(FragBF& f, const unsigned short* lds,
                                              int stride, int colbase, int kbase) {
    const int lane = threadIdx.x & 31;
    const int n  = colbase + (lane & 15);
    const int kb = kbase + ((lane >> 4) << 3);
#pragma unroll
    for (int v = 0; v < 8; ++v) {
        int k = kb + ((v < 4) ? (2 * v) : (16 + 2 * (v - 4)));
        unsigned int lo = lds[k * stride + n];
        unsigned int hi = lds[(k + 1) * stride + n];
        f.u[v] = lo | (hi << 16);
    }
}

__device__ __forceinline__ f32x8 wmma_bf16(const FragBF& a, const FragBF& b, f32x8 c) {
    return __builtin_amdgcn_wmma_f32_16x16x32_bf16(
        false, a.v, false, b.v, (short)0, c, false, false);
}

// ---------------------------------------------------------------------------
// fp32 -> bf16 conversion
// ---------------------------------------------------------------------------
__global__ void f2bf_kernel(const float* __restrict__ in,
                            unsigned short* __restrict__ out, int n) {
    int i = blockIdx.x * 256 + threadIdx.x;
    if (i < n) out[i] = f2bf(in[i]);
}

// ---------------------------------------------------------------------------
// GEMM: out[M,N] = A[M,K](bf16) @ W[N,K](bf16)^T + bias
// 128 threads = 4 waves. Block tile 128x64, wave tile 32x64, K-step 32.
// Double-buffered LDS, async global->LDS prefetch of tile k+1 during
// compute of tile k.
// ---------------------------------------------------------------------------
__global__ void __launch_bounds__(128)
gemm_bias_kernel(const unsigned short* __restrict__ A,
                 const unsigned short* __restrict__ W,
                 const float* __restrict__ bias,
                 float* __restrict__ outF,
                 unsigned short* __restrict__ outB,
                 int M, int N, int K, int store_bf16) {
    const int SA = 40;                           // LDS row stride (16B aligned)
    __shared__ unsigned short As[2][128 * SA];
    __shared__ unsigned short Bs[2][64 * SA];

    const int tid  = threadIdx.x;
    const int wave = tid >> 5;
    const int lane = tid & 31;
    const int m0 = blockIdx.y * 128;
    const int n0 = blockIdx.x * 64;

    f32x8 acc0[4] = {};
    f32x8 acc1[4] = {};

    // A: thread t stages its full row (32 bf16 = 4 x 16B). B: 2 threads/row.
    const int brow = tid >> 1;
    const int bseg = (tid & 1) * 16;

    auto stage = [&](int buf, int k0) {
        const unsigned short* ga = A + (size_t)(m0 + tid) * K + k0;
        unsigned short* la = As[buf] + tid * SA;
        cp16(la + 0,  ga + 0);
        cp16(la + 8,  ga + 8);
        cp16(la + 16, ga + 16);
        cp16(la + 24, ga + 24);
        const unsigned short* gb = W + (size_t)(n0 + brow) * K + k0 + bseg;
        unsigned short* lb = Bs[buf] + brow * SA + bseg;
        cp16(lb + 0, gb + 0);
        cp16(lb + 8, gb + 8);
    };

    stage(0, 0);
    int buf = 0;
    for (int k0 = 0; k0 < K; k0 += 32) {
        async_wait_all();
        __syncthreads();                         // buf ready; prev readers done
        if (k0 + 32 < K) stage(buf ^ 1, k0 + 32);

        FragBF a0, a1;
        load_frag_row(a0, As[buf], SA, wave * 32, 0);
        load_frag_row(a1, As[buf], SA, wave * 32 + 16, 0);
#pragma unroll
        for (int nt = 0; nt < 4; ++nt) {
            FragBF b;
            load_frag_row(b, Bs[buf], SA, nt * 16, 0);
            acc0[nt] = wmma_bf16(a0, b, acc0[nt]);
            acc1[nt] = wmma_bf16(a1, b, acc1[nt]);
        }
        buf ^= 1;
    }

    const int n    = lane & 15;
    const int half = lane >> 4;
#pragma unroll
    for (int nt = 0; nt < 4; ++nt) {
        int col = n0 + nt * 16 + n;
        float bv = bias[col];
#pragma unroll
        for (int r = 0; r < 8; ++r) {
            int mr0 = m0 + wave * 32 + r + 8 * half;
            int mr1 = mr0 + 16;
            float v0 = acc0[nt][r] + bv;
            float v1 = acc1[nt][r] + bv;
            if (store_bf16) {
                outB[(size_t)mr0 * N + col] = f2bf(v0);
                outB[(size_t)mr1 * N + col] = f2bf(v1);
            } else {
                outF[(size_t)mr0 * N + col] = v0;
                outF[(size_t)mr1 * N + col] = v1;
            }
        }
    }
}

// ---------------------------------------------------------------------------
// Flash attention. qkv: [B*S, 3072] bf16, head h occupies cols
// [h*192 .. h*192+191] as (q|k|v) of 64 each. vals: [B*S, 1024] bf16.
// Grid: (S/64, H, B). 128 threads = 4 waves; wave owns 16 query rows.
// ---------------------------------------------------------------------------
__global__ void __launch_bounds__(128)
attn_kernel(const unsigned short* __restrict__ qkv,
            unsigned short* __restrict__ vals) {
    const int SD = 72;                        // LDS stride (even, 16B aligned)
    __shared__ unsigned short Qs[64 * SD];
    __shared__ unsigned short Ks[64 * SD];
    __shared__ unsigned short Vs[64 * SD];
    __shared__ unsigned short Ps[4][16 * SD];

    const int tid  = threadIdx.x;
    const int wave = tid >> 5;
    const int lane = tid & 31;
    const int nlane = lane & 15;
    const int half  = lane >> 4;

    const int q0 = blockIdx.x * 64;
    const int h  = blockIdx.y;
    const int b  = blockIdx.z;
    const size_t rowbase = (size_t)b * SEQ;
    const int qcol = h * 192, kcol = qcol + 64, vcol = qcol + 128;

    const int row = tid >> 1;                 // 0..63
    const int seg = (tid & 1) * 32;           // half of HDIM, ushort units

    // ---- stage Q tile ----
    {
        const unsigned short* s = qkv + (rowbase + q0 + row) * QKVW + qcol + seg;
        unsigned short* d = Qs + row * SD + seg;
        cp16(d + 0, s + 0); cp16(d + 8,  s + 8);
        cp16(d + 16, s + 16); cp16(d + 24, s + 24);
    }
    async_wait_all();
    __syncthreads();
    FragBF aq0, aq1;
    load_frag_row(aq0, Qs, SD, wave * 16, 0);
    load_frag_row(aq1, Qs, SD, wave * 16, 32);

    f32x8 o[4] = {};
    float mrow[8], lrow[8];
#pragma unroll
    for (int r = 0; r < 8; ++r) { mrow[r] = -1e30f; lrow[r] = 0.0f; }
    const float scale = 0.125f;               // 1/sqrt(64)

    for (int kt = 0; kt < SEQ; kt += 64) {
        __syncthreads();                      // readers of Ks/Vs done
        {
            const unsigned short* s = qkv + (rowbase + kt + row) * QKVW + kcol + seg;
            unsigned short* d = Ks + row * SD + seg;
            cp16(d + 0, s + 0); cp16(d + 8, s + 8);
            cp16(d + 16, s + 16); cp16(d + 24, s + 24);
            const unsigned short* sv = qkv + (rowbase + kt + row) * QKVW + vcol + seg;
            unsigned short* dv = Vs + row * SD + seg;
            cp16(dv + 0, sv + 0); cp16(dv + 8, sv + 8);
            cp16(dv + 16, sv + 16); cp16(dv + 24, sv + 24);
        }
        async_wait_all();
        __syncthreads();

        // ---- S = Q @ K^T (16x64 per wave) ----
        f32x8 s4[4] = {};
#pragma unroll
        for (int nt = 0; nt < 4; ++nt) {
            FragBF bk;
            load_frag_row(bk, Ks, SD, nt * 16, 0);
            s4[nt] = wmma_bf16(aq0, bk, s4[nt]);
            load_frag_row(bk, Ks, SD, nt * 16, 32);
            s4[nt] = wmma_bf16(aq1, bk, s4[nt]);
        }

        // ---- online softmax (row stats live in 16-lane halves) ----
#pragma unroll
        for (int r = 0; r < 8; ++r) {
#pragma unroll
            for (int nt = 0; nt < 4; ++nt) s4[nt][r] *= scale;
            float mx = -1e30f;
#pragma unroll
            for (int nt = 0; nt < 4; ++nt) mx = fmaxf(mx, s4[nt][r]);
#pragma unroll
            for (int off = 1; off < 16; off <<= 1)
                mx = fmaxf(mx, __shfl_xor(mx, off, 32));
            float mnew  = fmaxf(mrow[r], mx);
            float alpha = __expf(mrow[r] - mnew);
            float rs = 0.0f;
#pragma unroll
            for (int nt = 0; nt < 4; ++nt) {
                float p = __expf(s4[nt][r] - mnew);
                s4[nt][r] = p;
                rs += p;
            }
#pragma unroll
            for (int off = 1; off < 16; off <<= 1)
                rs += __shfl_xor(rs, off, 32);
            lrow[r] = lrow[r] * alpha + rs;
            mrow[r] = mnew;
#pragma unroll
            for (int nt = 0; nt < 4; ++nt) o[nt][r] *= alpha;
        }

        // ---- P (bf16) to per-wave LDS, then O += P @ V ----
        unsigned short* pw = Ps[wave];
#pragma unroll
        for (int nt = 0; nt < 4; ++nt)
#pragma unroll
            for (int r = 0; r < 8; ++r)
                pw[(r + 8 * half) * SD + nt * 16 + nlane] = f2bf(s4[nt][r]);
        __syncthreads();

        FragBF ap;
        load_frag_row(ap, pw, SD, 0, 0);
#pragma unroll
        for (int nt = 0; nt < 4; ++nt) {
            FragBF bv;
            load_frag_col(bv, Vs, SD, nt * 16, 0);
            o[nt] = wmma_bf16(ap, bv, o[nt]);
        }
        load_frag_row(ap, pw, SD, 0, 32);
#pragma unroll
        for (int nt = 0; nt < 4; ++nt) {
            FragBF bv;
            load_frag_col(bv, Vs, SD, nt * 16, 32);
            o[nt] = wmma_bf16(ap, bv, o[nt]);
        }
    }

    // ---- normalize and write vals[b,s, h*64 + d] ----
#pragma unroll
    for (int r = 0; r < 8; ++r) {
        float inv = 1.0f / lrow[r];
        size_t mr = rowbase + q0 + wave * 16 + r + 8 * half;
#pragma unroll
        for (int nt = 0; nt < 4; ++nt)
            vals[mr * EMBED + h * HDIM + nt * 16 + nlane] = f2bf(o[nt][r] * inv);
    }
}

// ---------------------------------------------------------------------------
extern "C" void kernel_launch(void* const* d_in, const int* in_sizes, int n_in,
                              void* d_out, int out_size, void* d_ws, size_t ws_size,
                              hipStream_t stream) {
    (void)in_sizes; (void)n_in; (void)out_size; (void)ws_size;
    const float* x     = (const float*)d_in[0];
    const float* Wqkv  = (const float*)d_in[1];
    const float* bqkv  = (const float*)d_in[2];
    const float* Wo    = (const float*)d_in[3];
    const float* bo    = (const float*)d_in[4];
    float* out = (float*)d_out;

    unsigned short* x_bf    = (unsigned short*)d_ws;
    unsigned short* wqkv_bf = x_bf    + (size_t)ROWS * EMBED;
    unsigned short* wo_bf   = wqkv_bf + (size_t)QKVW * EMBED;
    unsigned short* qkv_bf  = wo_bf   + (size_t)EMBED * EMBED;
    unsigned short* vals_bf = qkv_bf  + (size_t)ROWS * QKVW;

    f2bf_kernel<<<(ROWS * EMBED) / 256, 256, 0, stream>>>(x, x_bf, ROWS * EMBED);
    f2bf_kernel<<<(QKVW * EMBED) / 256, 256, 0, stream>>>(Wqkv, wqkv_bf, QKVW * EMBED);
    f2bf_kernel<<<(EMBED * EMBED) / 256, 256, 0, stream>>>(Wo, wo_bf, EMBED * EMBED);

    gemm_bias_kernel<<<dim3(QKVW / 64, ROWS / 128), 128, 0, stream>>>(
        x_bf, wqkv_bf, bqkv, nullptr, qkv_bf, ROWS, QKVW, EMBED, 1);

    attn_kernel<<<dim3(SEQ / 64, HEADS, BATCH), 128, 0, stream>>>(qkv_bf, vals_bf);

    gemm_bias_kernel<<<dim3(EMBED / 64, ROWS / 128), 128, 0, stream>>>(
        vals_bf, wo_bf, bo, out, nullptr, ROWS, EMBED, EMBED, 0);
}
